// Cross_AgentAttention_3650722202270
// MI455X (gfx1250) — compile-verified, matrix operand order checked by reading, not run on hardware
//
#include <hip/hip_runtime.h>
#include <hip/hip_bf16.h>
#include <math.h>

// ---------------- problem constants ----------------
#define BATCH 8
#define CDIM  256
#define HDIM  64
#define WDIM  64
#define NTOK  (HDIM*WDIM)     // 4096
#define NH    8
#define HD    32              // head dim
#define NAGT  16              // agent tokens
#define SCALE 0.17677669529663689f   // 32^-0.5

typedef __attribute__((ext_vector_type(16))) _Float16 v16h;
typedef __attribute__((ext_vector_type(8)))  _Float16 v8h;
typedef __attribute__((ext_vector_type(8)))  float    v8f;

// K-offset (within a 32-wide K slice) of element j of a v16h A/B fragment,
// before adding the per-lane-half offset (+8 for lanes 16..31).
__device__ __forceinline__ int kmap(int j) { return (j < 8) ? j : (8 + j); }

__device__ __forceinline__ v8f wmma16(v16h a, v16h b, v8f c) {
  return __builtin_amdgcn_wmma_f32_16x16x32_f16(false, a, false, b, (short)0, c,
                                                false, false);
}

// load a v16h fragment whose per-lane elements are K = base+{0..7} and
// base+{16..23} (two contiguous 16-byte chunks)
__device__ __forceinline__ v16h ldfrag(const _Float16* __restrict__ p) {
  v8h lo = *(const v8h*)p;
  v8h hi = *(const v8h*)(p + 16);
  v16h r;
#pragma unroll
  for (int j = 0; j < 8; ++j) { r[j] = lo[j]; r[j + 8] = hi[j]; }
  return r;
}

// ---------------------------------------------------------------------------
// 0a) transpose+convert inputs: (B,C,N) f32 -> (B,N,C) f16, LDS 32x32 tiles
// grid(N/32, C/32, B*2) block(32,8)
// ---------------------------------------------------------------------------
__global__ void k_cvt_x(const float* __restrict__ in1,
                        const float* __restrict__ in2,
                        _Float16* __restrict__ xh1,
                        _Float16* __restrict__ xh2) {
  __shared__ float tile[32][33];
  int which = blockIdx.z & 1;
  int b = blockIdx.z >> 1;
  const float* src = which ? in2 : in1;
  _Float16* dst = which ? xh2 : xh1;
  int n0 = blockIdx.x * 32, c0 = blockIdx.y * 32;
  int tx = threadIdx.x, ty = threadIdx.y;
#pragma unroll
  for (int i = ty; i < 32; i += 8)
    tile[i][tx] = src[(size_t)b * CDIM * NTOK + (size_t)(c0 + i) * NTOK + n0 + tx];
  __syncthreads();
#pragma unroll
  for (int i = ty; i < 32; i += 8)
    dst[((size_t)b * NTOK + n0 + i) * CDIM + c0 + tx] = (_Float16)tile[tx][i];
}

// 0b) transposed f16 weights: kvwT[j][k] = kv_w[k][j]; projwT[j][k] = proj_w[k][j]
__global__ void k_cvt_w(const float* __restrict__ kv_w,
                        const float* __restrict__ proj_w,
                        _Float16* __restrict__ kvwT,
                        _Float16* __restrict__ projwT) {
  size_t idx = (size_t)blockIdx.x * blockDim.x + threadIdx.x;
  if (idx < (size_t)2 * CDIM * CDIM) {
    int j = (int)(idx / CDIM), k = (int)(idx % CDIM);
    kvwT[idx] = (_Float16)kv_w[(size_t)k * (2 * CDIM) + j];
  } else if (idx < (size_t)3 * CDIM * CDIM) {
    size_t i2 = idx - (size_t)2 * CDIM * CDIM;
    int j = (int)(i2 / CDIM), k = (int)(i2 % CDIM);
    projwT[i2] = (_Float16)proj_w[(size_t)k * CDIM + j];
  }
}

// ---------------------------------------------------------------------------
// 1) agent tokens: 16x16 block-mean of guidmap, then affine with q_w/q_b
// ---------------------------------------------------------------------------
__global__ void k_agent_pool(const float* __restrict__ guid,
                             const float* __restrict__ q_w,
                             const float* __restrict__ q_b,
                             float* __restrict__ agent) {
  __shared__ float red[256];
  int blk = blockIdx.x;
  int b = blk >> 4, a = blk & 15;
  int py = a >> 2, px = a & 3;
  int t = threadIdx.x;
  int y = py * 16 + (t >> 4);
  int x = px * 16 + (t & 15);
  red[t] = guid[(size_t)b * NTOK + y * WDIM + x];
  __syncthreads();
  for (int s = 128; s > 0; s >>= 1) {
    if (t < s) red[t] += red[t + s];
    __syncthreads();
  }
  float gmean = red[0] * (1.0f / 256.0f);
  int c = t;
  int h = c >> 5, d = c & 31;
  agent[(((size_t)b * NH + h) * NAGT + a) * HD + d] = gmean * q_w[c] + q_b[c];
}

// ---------------------------------------------------------------------------
// 2) positional biases (bilinear 4x4 -> 64x64, half-pixel, edge clamp)
// ---------------------------------------------------------------------------
__device__ __forceinline__ float bilin4(const float* __restrict__ t, float sy, float sx) {
  float fy = floorf(sy), fx = floorf(sx);
  int y0 = (int)fy, x0 = (int)fx;
  float wy = sy - fy, wx = sx - fx;
  int y0c = min(max(y0, 0), 3), y1c = min(max(y0 + 1, 0), 3);
  int x0c = min(max(x0, 0), 3), x1c = min(max(x0 + 1, 0), 3);
  float v00 = t[y0c * 4 + x0c], v01 = t[y0c * 4 + x1c];
  float v10 = t[y1c * 4 + x0c], v11 = t[y1c * 4 + x1c];
  return v00 * (1.f - wy) * (1.f - wx) + v01 * (1.f - wy) * wx +
         v10 * wy * (1.f - wx) + v11 * wy * wx;
}

__global__ void k_bias(const float* __restrict__ an_bias,
                       const float* __restrict__ na_bias,
                       const float* __restrict__ ah_bias,
                       const float* __restrict__ aw_bias,
                       const float* __restrict__ ha_bias,
                       const float* __restrict__ wa_bias,
                       float* __restrict__ posb,
                       float* __restrict__ agentb) {
  size_t idx = (size_t)blockIdx.x * blockDim.x + threadIdx.x;
  if (idx >= (size_t)NH * NAGT * NTOK) return;
  int n = idx % NTOK;
  int a = (idx / NTOK) % NAGT;
  int h = idx / (NTOK * NAGT);
  int y = n >> 6, x = n & 63;
  float sy = (y + 0.5f) * (4.0f / 64.0f) - 0.5f;
  float sx = (x + 0.5f) * (4.0f / 64.0f) - 0.5f;
  const float* an = an_bias + ((size_t)h * NAGT + a) * 16;
  const float* na = na_bias + ((size_t)h * NAGT + a) * 16;
  posb[((size_t)h * NAGT + a) * NTOK + n] =
      bilin4(an, sy, sx) + ah_bias[((size_t)h * NAGT + a) * HDIM + y] +
      aw_bias[((size_t)h * NAGT + a) * WDIM + x];
  agentb[((size_t)h * NTOK + n) * NAGT + a] =
      bilin4(na, sy, sx) + ha_bias[((size_t)h * HDIM + y) * NAGT + a] +
      wa_bias[((size_t)h * WDIM + x) * NAGT + a];
}

// ---------------------------------------------------------------------------
// 3) KV GEMM: kvh[b,n,0:512] = xh[b,n,:] @ kv_w + kv_b   (all-f16 fragments)
// grid(2048, 2, 2) block(32,4): wave owns a 16(M)x64(J) tile
// ---------------------------------------------------------------------------
__global__ void k_kv_gemm(const _Float16* __restrict__ xh1,
                          const _Float16* __restrict__ xh2,
                          const _Float16* __restrict__ kvwT,
                          const float* __restrict__ kv_b,
                          _Float16* __restrict__ kvh1,
                          _Float16* __restrict__ kvh2) {
  const _Float16* xh = blockIdx.z ? xh2 : xh1;
  _Float16* dst = blockIdx.z ? kvh2 : kvh1;
  int lane = threadIdx.x;
  int jbase = (blockIdx.y * 4 + threadIdx.y) * 64;
  size_t m0 = (size_t)blockIdx.x * 16;
  int row = lane & 15;
  int lhi = (lane & 16) ? 8 : 0;
  const _Float16* arow = xh + (m0 + row) * CDIM + lhi;

  v8f acc0 = {}, acc1 = {}, acc2 = {}, acc3 = {};
  for (int k = 0; k < CDIM; k += 32) {
    __builtin_prefetch(arow + k + 32, 0, 1);
    v16h af = ldfrag(arow + k);
    v16h bf0 = ldfrag(kvwT + (size_t)(jbase + 0 * 16 + row) * CDIM + k + lhi);
    v16h bf1 = ldfrag(kvwT + (size_t)(jbase + 1 * 16 + row) * CDIM + k + lhi);
    v16h bf2 = ldfrag(kvwT + (size_t)(jbase + 2 * 16 + row) * CDIM + k + lhi);
    v16h bf3 = ldfrag(kvwT + (size_t)(jbase + 3 * 16 + row) * CDIM + k + lhi);
    acc0 = wmma16(af, bf0, acc0);
    acc1 = wmma16(af, bf1, acc1);
    acc2 = wmma16(af, bf2, acc2);
    acc3 = wmma16(af, bf3, acc3);
  }
#pragma unroll
  for (int t = 0; t < 4; ++t) {
    int col = jbase + t * 16 + row;
    v8f acc = (t == 0) ? acc0 : (t == 1) ? acc1 : (t == 2) ? acc2 : acc3;
    float bias = kv_b[col];
#pragma unroll
    for (int v = 0; v < 8; ++v) {
      int m = v + lhi;
      dst[(m0 + m) * (2 * CDIM) + col] = (_Float16)(acc[v] + bias);
    }
  }
}

// ---------------------------------------------------------------------------
// 4a) agent scores: S[b,h,a,n] = scale*agent @ k^T + pos_bias
// grid(64, NH, B) block(32)
// ---------------------------------------------------------------------------
__global__ void k_agent_scores(const _Float16* __restrict__ kvh,  // k at +0
                               const float* __restrict__ agent,
                               const float* __restrict__ posb,
                               float* __restrict__ S) {
  int lane = threadIdx.x;
  int h = blockIdx.y, b = blockIdx.z;
  int row = lane & 15;
  int lhi = (lane & 16) ? 8 : 0;

  v16h af;
#pragma unroll
  for (int j = 0; j < 16; ++j) {
    int d = lhi + kmap(j);
    af[j] = (_Float16)(SCALE * agent[(((size_t)b * NH + h) * NAGT + row) * HD + d]);
  }
#pragma unroll
  for (int t = 0; t < 4; ++t) {
    int ncol = blockIdx.x * 64 + t * 16 + row;
    v16h bf = ldfrag(kvh + ((size_t)b * NTOK + ncol) * (2 * CDIM) + h * HD + lhi);
    v8f acc = {};
    acc = wmma16(af, bf, acc);
#pragma unroll
    for (int v = 0; v < 8; ++v) {
      int m = v + lhi;
      S[(((size_t)b * NH + h) * NAGT + m) * NTOK + ncol] =
          acc[v] + posb[((size_t)h * NAGT + m) * NTOK + ncol];
    }
  }
}

// ---------------------------------------------------------------------------
// 4b) softmax over n=4096 per (b,h,a) row; writes probabilities as f16
// grid(B*NH*NAGT) block(256)
// ---------------------------------------------------------------------------
__global__ void k_softmax_rows(const float* __restrict__ S,
                               _Float16* __restrict__ Ph) {
  __shared__ float red[256];
  int t = threadIdx.x;
  const float* rowp = S + (size_t)blockIdx.x * NTOK;
  _Float16* prow = Ph + (size_t)blockIdx.x * NTOK;
  float v[NTOK / 256];
  float lmax = -3.4e38f;
#pragma unroll
  for (int i = 0; i < NTOK / 256; ++i) {
    v[i] = rowp[i * 256 + t];
    lmax = fmaxf(lmax, v[i]);
  }
  red[t] = lmax; __syncthreads();
  for (int s = 128; s > 0; s >>= 1) {
    if (t < s) red[t] = fmaxf(red[t], red[t + s]);
    __syncthreads();
  }
  float mx = red[0]; __syncthreads();
  float ls = 0.f;
#pragma unroll
  for (int i = 0; i < NTOK / 256; ++i) { v[i] = __expf(v[i] - mx); ls += v[i]; }
  red[t] = ls; __syncthreads();
  for (int s = 128; s > 0; s >>= 1) {
    if (t < s) red[t] += red[t + s];
    __syncthreads();
  }
  float inv = 1.0f / red[0];
#pragma unroll
  for (int i = 0; i < NTOK / 256; ++i)
    prow[i * 256 + t] = (_Float16)(v[i] * inv);
}

// ---------------------------------------------------------------------------
// 4c) agent_v = P @ v  (M=16, N=32, K=4096 over 16 waves, atomic f32 reduce)
// grid(16, B*NH) block(32)
// ---------------------------------------------------------------------------
__global__ void k_pv(const _Float16* __restrict__ kvh,  // v at +CDIM
                     const _Float16* __restrict__ Ph,
                     float* __restrict__ agent_v) {
  int lane = threadIdx.x;
  int bh = blockIdx.y;
  int b = bh / NH, h = bh % NH;
  int row = lane & 15;
  int lhi = (lane & 16) ? 8 : 0;
  int kbase0 = blockIdx.x * (NTOK / 16);
  const _Float16* prow = Ph + (((size_t)b * NH + h) * NAGT + row) * NTOK + lhi;

  v8f acc0 = {}, acc1 = {};
  for (int kb = kbase0; kb < kbase0 + NTOK / 16; kb += 32) {
    v16h af = ldfrag(prow + kb);
    v16h bf0, bf1;
#pragma unroll
    for (int j = 0; j < 16; ++j) {
      int n = kb + lhi + kmap(j);
      const _Float16* vp = kvh + ((size_t)b * NTOK + n) * (2 * CDIM) + CDIM + h * HD;
      bf0[j] = vp[row];
      bf1[j] = vp[16 + row];
    }
    acc0 = wmma16(af, bf0, acc0);
    acc1 = wmma16(af, bf1, acc1);
  }
#pragma unroll
  for (int v = 0; v < 8; ++v) {
    int m = v + lhi;
    float* dst = agent_v + (((size_t)b * NH + h) * NAGT + m) * HD;
    atomicAdd(&dst[row], acc0[v]);
    atomicAdd(&dst[16 + row], acc1[v]);
  }
}

// ---------------------------------------------------------------------------
// 5) q-attention over agents + apply to agent_v1/2 + fused 3x3 depthwise conv
// one thread per (b,h,n); writes pre-projection activations (f16)
// ---------------------------------------------------------------------------
__global__ void __launch_bounds__(256)
k_qattn_out(const float* __restrict__ guid,
            const float* __restrict__ q_w,
            const float* __restrict__ q_b,
            const float* __restrict__ agent,
            const float* __restrict__ agentb,
            const float* __restrict__ av1,
            const float* __restrict__ av2,
            const _Float16* __restrict__ kvh1,
            const _Float16* __restrict__ kvh2,
            const float* __restrict__ dwc_w,
            const float* __restrict__ dwc_b,
            _Float16* __restrict__ pre1,
            _Float16* __restrict__ pre2) {
  size_t idx = (size_t)blockIdx.x * blockDim.x + threadIdx.x;
  if (idx >= (size_t)BATCH * NH * NTOK) return;
  int n = idx % NTOK;
  int h = (idx / NTOK) % NH;
  int b = idx / (NTOK * NH);

  float g = guid[(size_t)b * NTOK + n];
  float qv[HD];
#pragma unroll
  for (int d = 0; d < HD; ++d) {
    int c = h * HD + d;
    qv[d] = g * q_w[c] + q_b[c];
  }
  float s[NAGT];
  const float* ag = agent + ((size_t)b * NH + h) * NAGT * HD;
  float mx = -3.4e38f;
#pragma unroll
  for (int a = 0; a < NAGT; ++a) {
    float acc = 0.f;
#pragma unroll
    for (int d = 0; d < HD; ++d) acc += qv[d] * ag[a * HD + d];
    s[a] = acc * SCALE + agentb[((size_t)h * NTOK + n) * NAGT + a];
    mx = fmaxf(mx, s[a]);
  }
  float sum = 0.f;
#pragma unroll
  for (int a = 0; a < NAGT; ++a) { s[a] = __expf(s[a] - mx); sum += s[a]; }
  float inv = 1.0f / sum;
#pragma unroll
  for (int a = 0; a < NAGT; ++a) s[a] *= inv;

  float o1[HD], o2[HD];
#pragma unroll
  for (int d = 0; d < HD; ++d) { o1[d] = 0.f; o2[d] = 0.f; }
  const float* a1 = av1 + ((size_t)b * NH + h) * NAGT * HD;
  const float* a2 = av2 + ((size_t)b * NH + h) * NAGT * HD;
#pragma unroll
  for (int a = 0; a < NAGT; ++a) {
    float p = s[a];
#pragma unroll
    for (int d = 0; d < HD; ++d) {
      o1[d] += p * a1[a * HD + d];
      o2[d] += p * a2[a * HD + d];
    }
  }

  int y = n >> 6, x = n & 63;
#pragma unroll
  for (int d = 0; d < HD; ++d) {
    int c = h * HD + d;
    float dw1 = dwc_b[c], dw2 = dwc_b[c];
    const float* wc = dwc_w + (size_t)c * 9;
    for (int ky = -1; ky <= 1; ++ky) {
      int yy = y + ky;
      if (yy < 0 || yy >= HDIM) continue;
      for (int kx = -1; kx <= 1; ++kx) {
        int xx = x + kx;
        if (xx < 0 || xx >= WDIM) continue;
        float wv = wc[(ky + 1) * 3 + (kx + 1)];
        size_t nn = (size_t)b * NTOK + yy * WDIM + xx;
        dw1 += wv * (float)kvh1[nn * (2 * CDIM) + CDIM + c];
        dw2 += wv * (float)kvh2[nn * (2 * CDIM) + CDIM + c];
      }
    }
    pre1[((size_t)b * NTOK + n) * CDIM + c] = (_Float16)(o1[d] + dw1);
    pre2[((size_t)b * NTOK + n) * CDIM + c] = (_Float16)(o2[d] + dw2);
  }
}

// ---------------------------------------------------------------------------
// 6) projection GEMM + residual, transposed store to (b,c,n)
// grid(2048, 1, 2) block(32,4)
// ---------------------------------------------------------------------------
__global__ void k_proj_gemm(const _Float16* __restrict__ pre1,
                            const _Float16* __restrict__ pre2,
                            const _Float16* __restrict__ projwT,
                            const float* __restrict__ proj_b,
                            const float* __restrict__ in1,
                            const float* __restrict__ in2,
                            float* __restrict__ out) {
  const _Float16* pre = blockIdx.z ? pre2 : pre1;
  const float* res = blockIdx.z ? in2 : in1;
  float* dst = out + (size_t)blockIdx.z * BATCH * CDIM * NTOK;
  int lane = threadIdx.x;
  int jbase = threadIdx.y * 64;
  size_t m0 = (size_t)blockIdx.x * 16;
  int b = (int)(m0 / NTOK);
  int n0 = (int)(m0 % NTOK);
  int row = lane & 15;
  int lhi = (lane & 16) ? 8 : 0;
  const _Float16* arow = pre + (m0 + row) * CDIM + lhi;

  v8f acc0 = {}, acc1 = {}, acc2 = {}, acc3 = {};
  for (int k = 0; k < CDIM; k += 32) {
    __builtin_prefetch(arow + k + 32, 0, 1);
    v16h af = ldfrag(arow + k);
    v16h bf0 = ldfrag(projwT + (size_t)(jbase + 0 * 16 + row) * CDIM + k + lhi);
    v16h bf1 = ldfrag(projwT + (size_t)(jbase + 1 * 16 + row) * CDIM + k + lhi);
    v16h bf2 = ldfrag(projwT + (size_t)(jbase + 2 * 16 + row) * CDIM + k + lhi);
    v16h bf3 = ldfrag(projwT + (size_t)(jbase + 3 * 16 + row) * CDIM + k + lhi);
    acc0 = wmma16(af, bf0, acc0);
    acc1 = wmma16(af, bf1, acc1);
    acc2 = wmma16(af, bf2, acc2);
    acc3 = wmma16(af, bf3, acc3);
  }
#pragma unroll
  for (int t = 0; t < 4; ++t) {
    int col = jbase + t * 16 + row;
    v8f acc = (t == 0) ? acc0 : (t == 1) ? acc1 : (t == 2) ? acc2 : acc3;
    float bias = proj_b[col];
#pragma unroll
    for (int v = 0; v < 8; ++v) {
      int m = v + lhi;
      size_t o = (size_t)b * CDIM * NTOK + (size_t)col * NTOK + n0 + m;
      dst[o] = acc[v] + bias + res[o];
    }
  }
}

// ---------------------------------------------------------------------------
extern "C" void kernel_launch(void* const* d_in, const int* in_sizes, int n_in,
                              void* d_out, int out_size, void* d_ws, size_t ws_size,
                              hipStream_t stream) {
  const float* input1  = (const float*)d_in[0];
  const float* input2  = (const float*)d_in[1];
  const float* guidmap = (const float*)d_in[2];
  const float* kv_w    = (const float*)d_in[3];
  const float* kv_b    = (const float*)d_in[4];
  const float* q_w     = (const float*)d_in[5];
  const float* q_b     = (const float*)d_in[6];
  const float* proj_w  = (const float*)d_in[7];
  const float* proj_b  = (const float*)d_in[8];
  const float* dwc_w   = (const float*)d_in[9];
  const float* dwc_b   = (const float*)d_in[10];
  const float* an_bias = (const float*)d_in[11];
  const float* na_bias = (const float*)d_in[12];
  const float* ah_bias = (const float*)d_in[13];
  const float* aw_bias = (const float*)d_in[14];
  const float* ha_bias = (const float*)d_in[15];
  const float* wa_bias = (const float*)d_in[16];

  // ---- workspace carve (16B aligned; all sizes are multiples of 128 elems)
  char* base = (char*)d_ws;
  size_t off = 0;
  auto carveH = [&](size_t n) { _Float16* p = (_Float16*)(base + off); off += n * sizeof(_Float16); return p; };
  auto carveF = [&](size_t n) { float* p = (float*)(base + off); off += n * sizeof(float); return p; };

  _Float16* xh1    = carveH((size_t)BATCH * NTOK * CDIM);      // 8.39M halfs
  _Float16* xh2    = carveH((size_t)BATCH * NTOK * CDIM);
  _Float16* kvh1   = carveH((size_t)BATCH * NTOK * 2 * CDIM);  // 16.78M halfs
  _Float16* kvh2   = carveH((size_t)BATCH * NTOK * 2 * CDIM);
  _Float16* kvwT   = carveH((size_t)2 * CDIM * CDIM);
  _Float16* projwT = carveH((size_t)CDIM * CDIM);
  _Float16* Ph     = carveH((size_t)BATCH * NH * NAGT * NTOK); // 4.19M halfs
  _Float16* pre1   = carveH((size_t)BATCH * NTOK * CDIM);
  _Float16* pre2   = carveH((size_t)BATCH * NTOK * CDIM);
  float* agent  = carveF((size_t)BATCH * NH * NAGT * HD);
  float* av1    = carveF((size_t)BATCH * NH * NAGT * HD);
  float* av2    = carveF((size_t)BATCH * NH * NAGT * HD);
  float* posb   = carveF((size_t)NH * NAGT * NTOK);
  float* agentb = carveF((size_t)NH * NTOK * NAGT);
  float* S      = carveF((size_t)BATCH * NH * NAGT * NTOK);
  float* outp   = (float*)d_out;

  hipMemsetAsync(av1, 0, (size_t)2 * BATCH * NH * NAGT * HD * sizeof(float), stream);

  k_cvt_x<<<dim3(NTOK / 32, CDIM / 32, BATCH * 2), dim3(32, 8), 0, stream>>>(
      input1, input2, xh1, xh2);
  k_cvt_w<<<(3 * CDIM * CDIM + 255) / 256, 256, 0, stream>>>(kv_w, proj_w, kvwT, projwT);

  k_agent_pool<<<BATCH * NAGT, 256, 0, stream>>>(guidmap, q_w, q_b, agent);

  {
    size_t tot = (size_t)NH * NAGT * NTOK;
    k_bias<<<(int)((tot + 255) / 256), 256, 0, stream>>>(
        an_bias, na_bias, ah_bias, aw_bias, ha_bias, wa_bias, posb, agentb);
  }

  k_kv_gemm<<<dim3(BATCH * NTOK / 16, 2, 2), dim3(32, 4), 0, stream>>>(
      xh1, xh2, kvwT, kv_b, kvh1, kvh2);

  // attention 1
  k_agent_scores<<<dim3(NTOK / 64, NH, BATCH), 32, 0, stream>>>(kvh1, agent, posb, S);
  k_softmax_rows<<<BATCH * NH * NAGT, 256, 0, stream>>>(S, Ph);
  k_pv<<<dim3(16, BATCH * NH), 32, 0, stream>>>(kvh1, Ph, av1);

  // attention 2 (reuses S and Ph)
  k_agent_scores<<<dim3(NTOK / 64, NH, BATCH), 32, 0, stream>>>(kvh2, agent, posb, S);
  k_softmax_rows<<<BATCH * NH * NAGT, 256, 0, stream>>>(S, Ph);
  k_pv<<<dim3(16, BATCH * NH), 32, 0, stream>>>(kvh2, Ph, av2);

  {
    size_t tot = (size_t)BATCH * NH * NTOK;
    k_qattn_out<<<(int)((tot + 255) / 256), 256, 0, stream>>>(
        guidmap, q_w, q_b, agent, agentb, av1, av2, kvh1, kvh2, dwc_w, dwc_b,
        pre1, pre2);
  }

  k_proj_gemm<<<dim3(BATCH * NTOK / 16, 1, 2), dim3(32, 4), 0, stream>>>(
      pre1, pre2, projwT, proj_b, input1, input2, outp);
}
